// ChildSumTreeGRU_24739011625785
// MI455X (gfx1250) — compile-verified, matrix operand order checked by reading, not run on hardware
//
#include <hip/hip_runtime.h>

// ---------------------------------------------------------------------------
// ChildSum Tree-GRU on complete binary tree, T=128, DEPTH=11, N=2047, X=H=256
// bf16 WMMA (v_wmma_f32_16x16x32_bf16) for all matmuls, fp32 state/epilogues.
// ---------------------------------------------------------------------------

#define T_DIM   128
#define DEPTH_C 11
#define NNODES  2047
#define HDIM    256

typedef __bf16 bf16_t;
typedef __attribute__((ext_vector_type(16))) __bf16 v16bf;
typedef __attribute__((ext_vector_type(8)))  __bf16 v8bf;
typedef __attribute__((ext_vector_type(8)))  float  v8f;

// A-fragment (16x32 bf16) and B-fragment (32x16 bf16) share the same per-lane
// pattern: lane (l&15) owns matrix row (A: M-row of input, B: N-row of a
// row-major weight matrix), and half = l>>4 selects K sub-blocks:
//   half=0 -> K = kc+[0..7], kc+[16..23]
//   half=1 -> K = kc+[8..15], kc+[24..31]
__device__ __forceinline__ v16bf load_frag(const bf16_t* row, int kc, int half) {
  v8bf lo = *(const v8bf*)(row + kc + half * 8);
  v8bf hi = *(const v8bf*)(row + kc + 16 + half * 8);
  return __builtin_shufflevector(lo, hi, 0,1,2,3,4,5,6,7,8,9,10,11,12,13,14,15);
}

__device__ __forceinline__ v8f wmma_bf16(v16bf a, v16bf b, v8f c) {
  return __builtin_amdgcn_wmma_f32_16x16x32_bf16(
      /*neg_a=*/false, a, /*neg_b=*/false, b,
      /*c_mod=*/(short)0, c, /*reuse_a=*/false, /*reuse_b=*/false);
}

__device__ __forceinline__ float sigmoidf_(float x) {
  return 1.0f / (1.0f + __expf(-x));
}

// ---------------------------------------------------------------------------
// fp32 -> bf16 weight conversion (weights live in d_ws, L2-resident afterwards)
// ---------------------------------------------------------------------------
__global__ __launch_bounds__(256) void k_cvt_bf16(const float* __restrict__ src,
                                                  bf16_t* __restrict__ dst,
                                                  int n) {
  int i = blockIdx.x * 256 + threadIdx.x;
  if (i < n) dst[i] = (bf16_t)src[i];
}

// ---------------------------------------------------------------------------
// h = tanh(x @ W^T + bW) for all T*N rows.
// WG = 256 thr (8 waves). Stages 32 x-rows to LDS (bf16); each wave owns a
// 16(M) x 64(N) output strip: 4 accumulators, K-loop of 8 WMMA steps each.
// ---------------------------------------------------------------------------
__global__ __launch_bounds__(256) void k_proj(const float* __restrict__ x,
                                              const bf16_t* __restrict__ Wb,
                                              const float* __restrict__ bW,
                                              float* __restrict__ h) {
  __shared__ __align__(16) bf16_t xs[32][HDIM];
  const int tid = threadIdx.x;
  const size_t rowbase = (size_t)blockIdx.x * 32;

  // cooperative stage: 32 rows x 256 fp32 -> bf16 LDS (float4-coalesced)
  const float4* xv = (const float4*)(x + rowbase * HDIM);
#pragma unroll
  for (int i = 0; i < 8; ++i) {
    int c = i * 256 + tid;            // float4 chunk 0..2047
    float4 v = xv[c];
    bf16_t* d = &xs[0][0] + c * 4;
    d[0] = (bf16_t)v.x; d[1] = (bf16_t)v.y;
    d[2] = (bf16_t)v.z; d[3] = (bf16_t)v.w;
  }
  __syncthreads();

  const int lane = tid & 31, wave = tid >> 5;
  const int half = lane >> 4, lrow = lane & 15;
  const int mtile = wave >> 2;          // 0..1  -> rows [mtile*16, +16)
  const int jbase = (wave & 3) * 4;     // 4 consecutive N-tiles

  v8f acc0 = {}, acc1 = {}, acc2 = {}, acc3 = {};
  const bf16_t* arow = &xs[mtile * 16 + lrow][0];
  const bf16_t* w0 = Wb + (size_t)((jbase + 0) * 16 + lrow) * HDIM;
  const bf16_t* w1 = Wb + (size_t)((jbase + 1) * 16 + lrow) * HDIM;
  const bf16_t* w2 = Wb + (size_t)((jbase + 2) * 16 + lrow) * HDIM;
  const bf16_t* w3 = Wb + (size_t)((jbase + 3) * 16 + lrow) * HDIM;

#pragma unroll
  for (int kc = 0; kc < HDIM; kc += 32) {
    v16bf a  = load_frag(arow, kc, half);
    v16bf b0 = load_frag(w0, kc, half);
    v16bf b1 = load_frag(w1, kc, half);
    v16bf b2 = load_frag(w2, kc, half);
    v16bf b3 = load_frag(w3, kc, half);
    acc0 = wmma_bf16(a, b0, acc0);
    acc1 = wmma_bf16(a, b1, acc1);
    acc2 = wmma_bf16(a, b2, acc2);
    acc3 = wmma_bf16(a, b3, acc3);
  }

  // epilogue: D layout -> lane holds [m = half*8+v][n = tile*16 + lrow]
  const int n0 = jbase * 16 + lrow;
#pragma unroll
  for (int v = 0; v < 8; ++v) {
    int m = mtile * 16 + half * 8 + v;
    size_t orow = (rowbase + m) * HDIM;
    h[orow + n0 +  0] = tanhf(acc0[v] + bW[n0 +  0]);
    h[orow + n0 + 16] = tanhf(acc1[v] + bW[n0 + 16]);
    h[orow + n0 + 32] = tanhf(acc2[v] + bW[n0 + 32]);
    h[orow + n0 + 48] = tanhf(acc3[v] + bW[n0 + 48]);
  }
}

// ---------------------------------------------------------------------------
// One tree level. WG handles 16 (t,node) rows. Children of node i are rows
// 2i+1, 2i+2 (contiguous). Phase 1: r/z1/z2 (shared A+B fragments per K-step),
// sigmoid, gate-combine into LDS. Phase 2: candidate GEMM + tanh + blend.
// ---------------------------------------------------------------------------
__global__ __launch_bounds__(256) void k_level(float* __restrict__ h,
                                               const bf16_t* __restrict__ Urb,
                                               const float* __restrict__ br,
                                               const bf16_t* __restrict__ Ucb,
                                               const float* __restrict__ bc,
                                               const bf16_t* __restrict__ Uzb,
                                               const float* __restrict__ bz,
                                               int lshift, int nodebase) {
  __shared__ __align__(16) bf16_t sc1[16][HDIM];   // child 1 (bf16)
  __shared__ __align__(16) bf16_t sc2[16][HDIM];   // child 2 (bf16)
  __shared__ __align__(16) bf16_t ssum[16][HDIM];  // child sum (bf16)
  __shared__ __align__(16) bf16_t srs[16][HDIM];   // r * h_sum (bf16)
  __shared__ float szc[16][HDIM];                  // z1*c1 + z2*c2 (fp32)
  __shared__ float szg[16][HDIM];                  // z1 + z2 (fp32)

  const int tid = threadIdx.x;
  const int Lmask = (1 << lshift) - 1;
  const int tilebase = blockIdx.x * 16;

  // stage children: thread = column, loop rows (fully coalesced per row)
  {
    int col = tid;
#pragma unroll 4
    for (int row = 0; row < 16; ++row) {
      int g = tilebase + row;
      int t = g >> lshift;
      int node = nodebase + (g & Lmask);
      size_t c1row = ((size_t)t * NNODES + 2 * node + 1) * HDIM;
      float a = h[c1row + col];
      float b = h[c1row + HDIM + col];   // child2 row is contiguous
      sc1[row][col] = (bf16_t)a;
      sc2[row][col] = (bf16_t)b;
      ssum[row][col] = (bf16_t)(a + b);
    }
  }
  __syncthreads();

  const int lane = tid & 31, wave = tid >> 5;
  const int half = lane >> 4, lrow = lane & 15;

  // ---- phase 1: r, z1, z2 for this wave's two 16-col tiles ----
#pragma unroll
  for (int jj = 0; jj < 2; ++jj) {
    const int j = wave * 2 + jj;
    v8f ar = {}, az1 = {}, az2 = {};
    const bf16_t* urrow = Urb + (size_t)(j * 16 + lrow) * HDIM;
    const bf16_t* uzrow = Uzb + (size_t)(j * 16 + lrow) * HDIM;
#pragma unroll
    for (int kc = 0; kc < HDIM; kc += 32) {
      v16bf as = load_frag(&ssum[lrow][0], kc, half);
      v16bf a1 = load_frag(&sc1[lrow][0], kc, half);
      v16bf a2 = load_frag(&sc2[lrow][0], kc, half);
      v16bf bu = load_frag(urrow, kc, half);
      v16bf bzf = load_frag(uzrow, kc, half);
      ar  = wmma_bf16(as, bu, ar);
      az1 = wmma_bf16(a1, bzf, az1);
      az2 = wmma_bf16(a2, bzf, az2);
    }
#pragma unroll
    for (int v = 0; v < 8; ++v) {
      int m = half * 8 + v;
      int n = j * 16 + lrow;
      float rv  = sigmoidf_(ar[v]  + br[n]);
      float z1v = sigmoidf_(az1[v] + bz[n]);
      float z2v = sigmoidf_(az2[v] + bz[n]);
      float c1 = (float)sc1[m][n];
      float c2 = (float)sc2[m][n];
      srs[m][n] = (bf16_t)(rv * (c1 + c2));
      szc[m][n] = z1v * c1 + z2v * c2;
      szg[m][n] = z1v + z2v;
    }
  }
  __syncthreads();

  // ---- phase 2: h_cand = tanh((r*h_sum) @ Uc^T + bc); blend; store ----
#pragma unroll
  for (int jj = 0; jj < 2; ++jj) {
    const int j = wave * 2 + jj;
    v8f ac = {};
    const bf16_t* ucrow = Ucb + (size_t)(j * 16 + lrow) * HDIM;
#pragma unroll
    for (int kc = 0; kc < HDIM; kc += 32) {
      v16bf a = load_frag(&srs[lrow][0], kc, half);
      v16bf b = load_frag(ucrow, kc, half);
      ac = wmma_bf16(a, b, ac);
    }
#pragma unroll
    for (int v = 0; v < 8; ++v) {
      int m = half * 8 + v;
      int n = j * 16 + lrow;
      float hcand = tanhf(ac[v] + bc[n]);
      float hn = szc[m][n] + (1.0f - szg[m][n]) * hcand;
      int g = tilebase + m;
      int t = g >> lshift;
      int node = nodebase + (g & Lmask);
      h[((size_t)t * NNODES + node) * HDIM + n] = hn;
    }
  }
}

// ---------------------------------------------------------------------------
// launch
// ---------------------------------------------------------------------------
extern "C" void kernel_launch(void* const* d_in, const int* in_sizes, int n_in,
                              void* d_out, int out_size, void* d_ws, size_t ws_size,
                              hipStream_t stream) {
  const float* x  = (const float*)d_in[0];
  const float* W  = (const float*)d_in[1];
  const float* bW = (const float*)d_in[2];
  const float* Ur = (const float*)d_in[3];
  const float* br = (const float*)d_in[4];
  const float* Uc = (const float*)d_in[5];
  const float* bc = (const float*)d_in[6];
  const float* Uz = (const float*)d_in[7];
  const float* bz = (const float*)d_in[8];
  float* h = (float*)d_out;

  const int WSZ = HDIM * HDIM;  // 65536 elements per weight matrix
  bf16_t* Wb  = (bf16_t*)d_ws;
  bf16_t* Urb = Wb  + WSZ;
  bf16_t* Ucb = Urb + WSZ;
  bf16_t* Uzb = Ucb + WSZ;

  k_cvt_bf16<<<WSZ / 256, 256, 0, stream>>>(W,  Wb,  WSZ);
  k_cvt_bf16<<<WSZ / 256, 256, 0, stream>>>(Ur, Urb, WSZ);
  k_cvt_bf16<<<WSZ / 256, 256, 0, stream>>>(Uc, Ucb, WSZ);
  k_cvt_bf16<<<WSZ / 256, 256, 0, stream>>>(Uz, Uzb, WSZ);

  // initial projection over all T*N rows (262016 = 32 * 8188)
  k_proj<<<(T_DIM * NNODES) / 32, 256, 0, stream>>>(x, Wb, bW, h);

  // bottom-up level sweep (sequential dependency between levels)
  for (int l = DEPTH_C - 2; l >= 0; --l) {
    int Lnodes = 1 << l;
    int blocks = (T_DIM * Lnodes) / 16;   // T=128 -> always exact
    k_level<<<blocks, 256, 0, stream>>>(h, Urb, br, Ucb, bc, Uzb, bz,
                                        l, Lnodes - 1);
  }
}
